// GraphVAE_65249143160984
// MI455X (gfx1250) — matrix operand order repaired and checked by reading, not compiled, not run on hardware
//
#include <hip/hip_runtime.h>
#include <hip/hip_bf16.h>
#include <math.h>

typedef __attribute__((ext_vector_type(16))) __bf16 v16bf;
typedef __attribute__((ext_vector_type(8)))  float  v8f;
typedef __attribute__((ext_vector_type(8)))  unsigned int v8u;
typedef unsigned short ushort_t;

#define BGRAPHS 128
#define NNODES  65536
#define NEDGES  131072
#define GLOBD   693
#define GPAD    704
#define LN_EPS  1e-5f
#define AST     40      // sA row stride (ushorts): 80B -> 16B aligned, conflict-free

#if __has_builtin(__builtin_amdgcn_global_load_async_to_lds_b128) && \
    __has_builtin(__builtin_amdgcn_s_wait_asynccnt)
#define HAVE_ASYNC_LDS 1
#endif

// ---------- helpers ----------

__device__ __forceinline__ ushort_t f2bfu(float f) {
  unsigned u = __builtin_bit_cast(unsigned, f);
  u += 0x7FFFu + ((u >> 16) & 1u);          // round-to-nearest-even
  return (ushort_t)(u >> 16);
}

__device__ __forceinline__ unsigned packbf(float a, float b) {
#if __has_builtin(__builtin_amdgcn_cvt_pk_bf16_f32)
  typedef __attribute__((ext_vector_type(2))) __bf16 v2bf;
  v2bf r = __builtin_amdgcn_cvt_pk_bf16_f32(a, b);
  return __builtin_bit_cast(unsigned, r);
#else
  return (unsigned)f2bfu(a) | ((unsigned)f2bfu(b) << 16);
#endif
}

__device__ __forceinline__ v8f wmma_bf16(v16bf a, v16bf b, v8f c) {
  return __builtin_amdgcn_wmma_f32_16x16x32_bf16(false, a, false, b, (short)0, c, false, false);
}

// A-operand (16x32 bf16) per-lane K base for VGPR p; hi = lane>=16
__device__ __forceinline__ int kbase_of(int p, int hi) {
  return ((p < 4) ? (2 * p) : (2 * p + 8)) + (hi ? 8 : 0);
}

// stage 16 consecutive f32 -> bf16 into LDS (dst 16B aligned, src 16B aligned)
__device__ __forceinline__ void stage16_f32(ushort_t* dst, const float* src) {
  const float4 q0 = *reinterpret_cast<const float4*>(src);
  const float4 q1 = *reinterpret_cast<const float4*>(src + 4);
  const float4 q2 = *reinterpret_cast<const float4*>(src + 8);
  const float4 q3 = *reinterpret_cast<const float4*>(src + 12);
  uint4 o0, o1;
  o0.x = packbf(q0.x, q0.y); o0.y = packbf(q0.z, q0.w);
  o0.z = packbf(q1.x, q1.y); o0.w = packbf(q1.z, q1.w);
  o1.x = packbf(q2.x, q2.y); o1.y = packbf(q2.z, q2.w);
  o1.z = packbf(q3.x, q3.y); o1.w = packbf(q3.z, q3.w);
  *reinterpret_cast<uint4*>(dst)     = o0;
  *reinterpret_cast<uint4*>(dst + 8) = o1;
}

// stage 16 pre-converted bf16 (src/dst 16B aligned); async LDS path when available
#if defined(HAVE_ASYNC_LDS)
typedef int v4i_ __attribute__((vector_size(16)));
typedef __attribute__((address_space(1))) v4i_* as1_v4i;
typedef __attribute__((address_space(3))) v4i_* as3_v4i;
#endif

__device__ __forceinline__ void stage16_bf16(ushort_t* dst, const ushort_t* src) {
#if defined(HAVE_ASYNC_LDS)
  __builtin_amdgcn_global_load_async_to_lds_b128(
      (as1_v4i)(v4i_*)(void*)const_cast<ushort_t*>(src),
      (as3_v4i)(v4i_*)(void*)dst, 0, 0);
  __builtin_amdgcn_global_load_async_to_lds_b128(
      (as1_v4i)(v4i_*)(void*)const_cast<ushort_t*>(src + 8),
      (as3_v4i)(v4i_*)(void*)(dst + 8), 0, 0);
#else
  *reinterpret_cast<uint4*>(dst)     = *reinterpret_cast<const uint4*>(src);
  *reinterpret_cast<uint4*>(dst + 8) = *reinterpret_cast<const uint4*>(src + 8);
#endif
}

// ---------- prep kernels ----------

__global__ void bn_scale_shift(const float* __restrict__ g, const float* __restrict__ b,
                               const float* __restrict__ mean, const float* __restrict__ var,
                               int din, float* __restrict__ scale, float* __restrict__ shift) {
  int k = blockIdx.x * blockDim.x + threadIdx.x;
  if (k < din) {
    float s = g[k] * rsqrtf(var[k] + LN_EPS);
    scale[k] = s;
    shift[k] = b[k] - mean[k] * s;
  }
}

// b1f = b1 + shift . w1 ; b1ones additionally folds x==1 over k in [gstart,din)
__global__ __launch_bounds__(128)
void bias_fold(const float* __restrict__ w1, const float* __restrict__ b1,
               const float* __restrict__ scale, const float* __restrict__ shift,
               int din, int dout, int gstart,
               float* __restrict__ b1f, float* __restrict__ b1ones) {
  __shared__ float r1[128], r2[128];
  const int j = blockIdx.x;
  const int t = threadIdx.x;
  float s1 = 0.f, s2 = 0.f;
  for (int k = t; k < din; k += 128) {
    const float w = w1[(size_t)k * dout + j];
    s1 += shift[k] * w;
    if (k >= gstart) s2 += scale[k] * w;
  }
  r1[t] = s1; r2[t] = s2;
  __syncthreads();
  for (int o = 64; o > 0; o >>= 1) {
    if (t < o) { r1[t] += r1[t + o]; r2[t] += r2[t + o]; }
    __syncthreads();
  }
  if (t == 0) {
    b1f[j]    = b1[j] + r1[0];
    b1ones[j] = b1[j] + r1[0] + r2[0];
  }
}

// Pre-swizzle weights [din x dout] (optionally scaled per-k) into WMMA B-fragment
// order: dst[((kc*NT+nt)*32+lane)*16 + e] with k = kc*32 + (e&8)*2 + (lane>>4)*8 + (e&7), n = nt*16 + (lane&15)
__global__ void swizzleB(const float* __restrict__ src, const float* __restrict__ scale,
                         int din, int dout, int NT, long total, ushort_t* __restrict__ dst) {
  long idx = (long)blockIdx.x * 256 + threadIdx.x;
  if (idx >= total) return;
  const int e    = (int)(idx & 15);
  const int lane = (int)((idx >> 4) & 31);
  const long blk = idx >> 9;
  const int nt = (int)(blk % NT);
  const int kc = (int)(blk / NT);
  const int k = kc * 32 + (e & 8) * 2 + (lane >> 4) * 8 + (e & 7);
  const int n = nt * 16 + (lane & 15);
  float v = 0.f;
  if (k < din && n < dout) {
    v = src[(size_t)k * dout + n];
    if (scale) v *= scale[k];
  }
  dst[idx] = f2bfu(v);
}

// f32 [rows x cols] -> bf16 [rows x dstStride] zero-padded
__global__ void f32_to_bf16_pad(const float* __restrict__ src, int cols,
                                ushort_t* __restrict__ dst, int dstStride, long total) {
  long idx = (long)blockIdx.x * 256 + threadIdx.x;
  if (idx >= total) return;
  const int row = (int)(idx / dstStride), col = (int)(idx % dstStride);
  dst[idx] = (col < cols) ? f2bfu(src[(size_t)row * cols + col]) : (ushort_t)0;
}

// ---------- fused row MLP (edge MODE 0 / node MODE 1), dout = 64 ----------
// 128 threads = 4 wave32, 64 rows per workgroup, double-buffered 32-wide K chunks.

template <int MODE>
__global__ __launch_bounds__(128)
void mlp64_fused(int KC,
                 const float* __restrict__ x0,        // edges/e1 (mode0) or agg (mode1)
                 const float* __restrict__ nds,       // node features
                 const int* __restrict__ idxS, const int* __restrict__ idxR,
                 const int* __restrict__ gid,
                 const ushort_t* __restrict__ gbf,    // padded bf16 globals [B x GPAD], null if folded
                 const ushort_t* __restrict__ W1sw, const float* __restrict__ b1,
                 const ushort_t* __restrict__ W2sw, const float* __restrict__ b2,
                 const float* __restrict__ lng, const float* __restrict__ lnb,
                 float* __restrict__ out) {
  __shared__ __align__(16) ushort_t sA[2][64 * AST];
  __shared__ __align__(16) ushort_t sH[64 * 68];
  __shared__ __align__(16) float    sF[64 * 68];
  __shared__ float sL1[128], sL2[128];

  const int t = threadIdx.x;
  const int lane = t & 31;
  const int wave = t >> 5;
  const int hi = lane >> 4;
  const int rowBase = blockIdx.x * 64;

  const int srow = t >> 1;        // staging row 0..63 (2 threads/row)
  const int half = t & 1;         // which 16-wide K half
  const int gr = rowBase + srow;
  int si = 0, ri = 0;
  if (MODE == 0) { si = idxS[gr]; ri = idxR[gr]; }
  const int gi = gid[gr];

  v8f acc[4];
  acc[0] = {0,0,0,0,0,0,0,0}; acc[1] = {0,0,0,0,0,0,0,0};
  acc[2] = {0,0,0,0,0,0,0,0}; acc[3] = {0,0,0,0,0,0,0,0};

  const int rA = wave * 16 + (lane & 15);
  const v8u* B1 = reinterpret_cast<const v8u*>(W1sw);
  const v8u* B2 = reinterpret_cast<const v8u*>(W2sw);
  ushort_t* const sdst0 = &sA[0][srow * AST + half * 16];
  ushort_t* const sdst1 = &sA[1][srow * AST + half * 16];

  auto do_chunk = [&](int kc) {
#if defined(HAVE_ASYNC_LDS)
    __builtin_amdgcn_s_wait_asynccnt(0);
#endif
    __syncthreads();
    const ushort_t* sb = &sA[kc & 1][0];
    v8u au;
#pragma unroll
    for (int p = 0; p < 8; ++p)
      au[p] = *reinterpret_cast<const unsigned int*>(&sb[rA * AST + kbase_of(p, hi)]);
    const v16bf af = __builtin_bit_cast(v16bf, au);
#pragma unroll
    for (int nt = 0; nt < 4; ++nt) {
      const v16bf bf = __builtin_bit_cast(v16bf, B1[((size_t)kc * 4 + nt) * 32 + lane]);
      acc[nt] = wmma_bf16(af, bf, acc[nt]);
    }
  };

  int kc = 0;
  {
    const float* p = x0 + (size_t)gr * 64 + half * 16;
#pragma unroll
    for (int s = 0; s < 2; ++s, ++kc) {
      stage16_f32((kc & 1) ? sdst1 : sdst0, p + s * 32);
      do_chunk(kc);
    }
  }
  if (MODE == 0) {
    const float* p = nds + (size_t)si * 64 + half * 16;
#pragma unroll
    for (int s = 0; s < 2; ++s, ++kc) {
      stage16_f32((kc & 1) ? sdst1 : sdst0, p + s * 32);
      do_chunk(kc);
    }
    const float* q = nds + (size_t)ri * 64 + half * 16;
#pragma unroll
    for (int s = 0; s < 2; ++s, ++kc) {
      stage16_f32((kc & 1) ? sdst1 : sdst0, q + s * 32);
      do_chunk(kc);
    }
  } else {
    const float* p = nds + (size_t)gr * 64 + half * 16;
#pragma unroll
    for (int s = 0; s < 2; ++s, ++kc) {
      stage16_f32((kc & 1) ? sdst1 : sdst0, p + s * 32);
      do_chunk(kc);
    }
  }
  {
    // globals slab, pre-converted bf16, zero-padded (no tail guard needed)
    const int kc0 = kc;
    const ushort_t* p = gbf + (size_t)gi * GPAD + half * 16;
    for (; kc < KC; ++kc) {
      stage16_bf16((kc & 1) ? sdst1 : sdst0, p + (kc - kc0) * 32);
      do_chunk(kc);
    }
  }
  __syncthreads();   // protect sA (unused later) / order before sH writes

  // bias + ReLU -> sH (bf16); C-layout: VGPR i -> row i + 8*hi, col = lane&15
  {
    const int cb = lane & 15;
#pragma unroll
    for (int nt = 0; nt < 4; ++nt) {
      const int col = nt * 16 + cb;
      const float bb = b1[col];
#pragma unroll
      for (int i = 0; i < 8; ++i) {
        const int r = wave * 16 + i + (hi << 3);
        float v = acc[nt][i] + bb;
        sH[r * 68 + col] = f2bfu(v > 0.f ? v : 0.f);
      }
    }
  }
  __syncthreads();

  // layer 2: K = 64 (2 chunks)
  v8f d2[4];
  d2[0] = {0,0,0,0,0,0,0,0}; d2[1] = {0,0,0,0,0,0,0,0};
  d2[2] = {0,0,0,0,0,0,0,0}; d2[3] = {0,0,0,0,0,0,0,0};
#pragma unroll
  for (int k2 = 0; k2 < 2; ++k2) {
    v8u au;
#pragma unroll
    for (int p = 0; p < 8; ++p)
      au[p] = *reinterpret_cast<const unsigned int*>(&sH[rA * 68 + k2 * 32 + kbase_of(p, hi)]);
    const v16bf af = __builtin_bit_cast(v16bf, au);
#pragma unroll
    for (int nt = 0; nt < 4; ++nt) {
      const v16bf bf = __builtin_bit_cast(v16bf, B2[((size_t)k2 * 4 + nt) * 32 + lane]);
      d2[nt] = wmma_bf16(af, bf, d2[nt]);
    }
  }
  {
    const int cb = lane & 15;
#pragma unroll
    for (int nt = 0; nt < 4; ++nt) {
      const int col = nt * 16 + cb;
      const float bb = b2[col];
#pragma unroll
      for (int i = 0; i < 8; ++i) {
        const int r = wave * 16 + i + (hi << 3);
        float v = d2[nt][i] + bb;
        sF[r * 68 + col] = v > 0.f ? v : 0.f;
      }
    }
  }
  __syncthreads();

  // per-row LayerNorm: 2 threads per row
  {
    const int row = t >> 1;
    const int co = (t & 1) * 32;
    const float* rp = &sF[row * 68 + co];
    float s = 0.f, ss = 0.f;
#pragma unroll 8
    for (int c = 0; c < 32; ++c) { const float v = rp[c]; s += v; ss += v * v; }
    sL1[t] = s; sL2[t] = ss;
    __syncthreads();
    const float S  = sL1[row * 2] + sL1[row * 2 + 1];
    const float SS = sL2[row * 2] + sL2[row * 2 + 1];
    const float mu = S * (1.f / 64.f);
    float var = SS * (1.f / 64.f) - mu * mu;
    var = var < 0.f ? 0.f : var;
    const float rstd = rsqrtf(var + LN_EPS);
    float* op = &out[(size_t)(rowBase + row) * 64 + co];
#pragma unroll 8
    for (int c = 0; c < 32; ++c) op[c] = (rp[c] - mu) * rstd * lng[co + c] + lnb[co + c];
  }
}

// ---------- generic WMMA GEMM for global MLP (rows=128) ----------

__global__ __launch_bounds__(32)
void gemm_bf16(const ushort_t* __restrict__ A, int Apad, int KC,
               const ushort_t* __restrict__ Bsw, int NT,
               const float* __restrict__ bias, int dN, int rows,
               float* __restrict__ outF, int outFs,
               ushort_t* __restrict__ outBF, int outBFs) {
  const int lane = threadIdx.x & 31;
  const int hi = lane >> 4;
  const int rt = blockIdx.x;
  const int nt = blockIdx.y;
  const int rA = rt * 16 + (lane & 15);
  const v8u* Bv = reinterpret_cast<const v8u*>(Bsw);
  v8f acc = {0,0,0,0,0,0,0,0};
  for (int kc = 0; kc < KC; ++kc) {
    v8u au;
#pragma unroll
    for (int p = 0; p < 8; ++p)
      au[p] = *reinterpret_cast<const unsigned int*>(&A[(size_t)rA * Apad + kc * 32 + kbase_of(p, hi)]);
    const v16bf af = __builtin_bit_cast(v16bf, au);
    const v16bf bf = __builtin_bit_cast(v16bf, Bv[((size_t)kc * NT + nt) * 32 + lane]);
    acc = wmma_bf16(af, bf, acc);
  }
  const int col = nt * 16 + (lane & 15);
  if (col >= dN) return;
  const float bb = bias[col];
#pragma unroll
  for (int i = 0; i < 8; ++i) {
    const int r = rt * 16 + i + (hi << 3);
    if (r < rows) {
      float v = acc[i] + bb;
      v = v > 0.f ? v : 0.f;
      if (outF)  outF [(size_t)r * outFs  + col] = v;
      if (outBF) outBF[(size_t)r * outBFs + col] = f2bfu(v);
    }
  }
}

__global__ __launch_bounds__(256)
void ln_glob(const float* __restrict__ x, int xstride,
             const float* __restrict__ g, const float* __restrict__ b,
             float* __restrict__ out, int D) {
  __shared__ float r1[256], r2[256];
  const int row = blockIdx.x;
  const int t = threadIdx.x;
  const float* xp = x + (size_t)row * xstride;
  float s = 0.f, ss = 0.f;
  for (int c = t; c < D; c += 256) { float v = xp[c]; s += v; ss += v * v; }
  r1[t] = s; r2[t] = ss;
  __syncthreads();
  for (int o = 128; o > 0; o >>= 1) {
    if (t < o) { r1[t] += r1[t + o]; r2[t] += r2[t + o]; }
    __syncthreads();
  }
  const float mu = r1[0] / (float)D;
  float var = r2[0] / (float)D - mu * mu;
  var = var < 0.f ? 0.f : var;
  const float rstd = rsqrtf(var + LN_EPS);
  float* op = out + (size_t)row * D;
  for (int c = t; c < D; c += 256) op[c] = (xp[c] - mu) * rstd * g[c] + b[c];
}

// ---------- scatter / segment sums / misc ----------

__global__ __launch_bounds__(256)
void scatter_add64(const float* __restrict__ e, const int* __restrict__ recv,
                   float* __restrict__ agg) {
  const size_t idx = (size_t)blockIdx.x * 256 + threadIdx.x;
  const int r = (int)(idx >> 6);
  const int f = (int)(idx & 63);
  unsafeAtomicAdd(&agg[(size_t)recv[r] * 64 + f], e[idx]);
}

// sorted-gid segment sum, 64 features, run-length accumulation -> few atomics
__global__ __launch_bounds__(64)
void segsum64(const float* __restrict__ x, const int* __restrict__ gid,
              int rows, float* __restrict__ out, int ostride) {
  const int f = threadIdx.x;
  const int r0 = blockIdx.x * 1024;
  int r1 = r0 + 1024; if (r1 > rows) r1 = rows;
  int cur = gid[r0];
  float acc = 0.f;
  for (int r = r0; r < r1; ++r) {
    const int g = gid[r];
    if (g != cur) { unsafeAtomicAdd(&out[(size_t)cur * ostride + f], acc); acc = 0.f; cur = g; }
    acc += x[(size_t)r * 64 + f];
  }
  unsafeAtomicAdd(&out[(size_t)cur * ostride + f], acc);
}

__global__ void fill_f32(float* __restrict__ p, float v, long n) {
  long i = (long)blockIdx.x * 256 + threadIdx.x;
  if (i < n) p[i] = v;
}

// build bf16 global-MLP input [B x 832] = [ge(64)|gn(64)|glb(693)|pad]
__global__ void build_gsin_bf(const float* __restrict__ gsin, const float* __restrict__ glb,
                              ushort_t* __restrict__ dst) {
  int idx = blockIdx.x * 256 + threadIdx.x;
  if (idx >= BGRAPHS * 832) return;
  const int row = idx / 832, col = idx % 832;
  float v = 0.f;
  if (col < 128) v = gsin[idx];
  else if (glb != nullptr && col < 821) v = glb[(size_t)row * GLOBD + (col - 128)];
  dst[idx] = f2bfu(v);
}

__global__ void sigmoid_col0(float* __restrict__ e2) {
  int idx = blockIdx.x * 256 + threadIdx.x;
  if (idx < NEDGES) {
    float v = e2[(size_t)idx * 64];
    e2[(size_t)idx * 64] = 1.f / (1.f + expf(-v));
  }
}

// ---------- host orchestration ----------

struct MlpP { const float *b1,*b2,*bn_b,*bn_g,*bn_mean,*bn_var,*ln_b,*ln_g,*w1,*w2; };
struct MlpW { float *scale,*shift,*b1f,*b1ones; ushort_t *W1sw,*W2sw;
              int din,dout,KC1,KC2,NT,gstart; };

static MlpP getMlp(void* const* d_in, int base) {
  MlpP m;
  m.b1      = (const float*)d_in[base + 0];
  m.b2      = (const float*)d_in[base + 1];
  m.bn_b    = (const float*)d_in[base + 2];
  m.bn_g    = (const float*)d_in[base + 3];
  m.bn_mean = (const float*)d_in[base + 4];
  m.bn_var  = (const float*)d_in[base + 5];
  m.ln_b    = (const float*)d_in[base + 6];
  m.ln_g    = (const float*)d_in[base + 7];
  m.w1      = (const float*)d_in[base + 8];
  m.w2      = (const float*)d_in[base + 9];
  return m;
}

static void prep_mlp(const MlpP& p, const MlpW& w, hipStream_t s) {
  bn_scale_shift<<<(w.din + 255) / 256, 256, 0, s>>>(p.bn_g, p.bn_b, p.bn_mean, p.bn_var,
                                                     w.din, w.scale, w.shift);
  bias_fold<<<w.dout, 128, 0, s>>>(p.w1, p.b1, w.scale, w.shift, w.din, w.dout, w.gstart,
                                   w.b1f, w.b1ones);
  long t1 = (long)w.KC1 * w.NT * 512;
  swizzleB<<<(t1 + 255) / 256, 256, 0, s>>>(p.w1, w.scale, w.din, w.dout, w.NT, t1, w.W1sw);
  long t2 = (long)w.KC2 * w.NT * 512;
  swizzleB<<<(t2 + 255) / 256, 256, 0, s>>>(p.w2, nullptr, w.dout, w.dout, w.NT, t2, w.W2sw);
}

extern "C" void kernel_launch(void* const* d_in, const int* in_sizes, int n_in,
                              void* d_out, int out_size, void* d_ws, size_t ws_size,
                              hipStream_t stream) {
  (void)in_sizes; (void)n_in; (void)out_size; (void)ws_size;
  const float* nodes     = (const float*)d_in[0];
  const float* edges     = (const float*)d_in[1];
  const int*   senders   = (const int*)d_in[2];
  const int*   receivers = (const int*)d_in[3];
  const int*   node_gid  = (const int*)d_in[4];
  const int*   edge_gid  = (const int*)d_in[5];
  // pytree-flattened params: per-MLP keys sorted [b1,b2,bn_b,bn_g,bn_mean,bn_var,ln_b,ln_g,w1,w2],
  // MLPs sorted [edge, glob, node]
  MlpP pp[6] = { getMlp(d_in, 6),  getMlp(d_in, 16), getMlp(d_in, 26),
                 getMlp(d_in, 36), getMlp(d_in, 46), getMlp(d_in, 56) };

  char* base = (char*)d_ws;
  size_t off = 0;
  auto cv = [&](size_t bytes) -> void* {
    void* p = base + off; off = (off + bytes + 255) & ~(size_t)255; return p; };

  const int dins[3]  = {885, 821, 821};   // edge, glob, node
  const int douts[3] = {64, 693, 64};
  const int gst[3]   = {192, 128, 128};   // start of the (ones-foldable) globals slab
  MlpW w[6];
  for (int i = 0; i < 6; ++i) {
    const int kind = i % 3;
    MlpW& m = w[i];
    m.din = dins[kind]; m.dout = douts[kind]; m.gstart = gst[kind];
    m.KC1 = (m.din + 31) / 32;
    m.KC2 = (m.dout + 31) / 32;
    m.NT  = (m.dout + 15) / 16;
    m.scale  = (float*)cv((size_t)m.din * 4);
    m.shift  = (float*)cv((size_t)m.din * 4);
    m.b1f    = (float*)cv((size_t)m.dout * 4);
    m.b1ones = (float*)cv((size_t)m.dout * 4);
    m.W1sw = (ushort_t*)cv((size_t)m.KC1 * m.NT * 512 * 2);
    m.W2sw = (ushort_t*)cv((size_t)m.KC2 * m.NT * 512 * 2);
  }
  float*    ebuf   = (float*)cv((size_t)NEDGES * 64 * 4);
  float*    nbuf   = (float*)cv((size_t)NNODES * 64 * 4);
  float*    gbuf   = (float*)cv((size_t)BGRAPHS * GLOBD * 4);
  ushort_t* gbf    = (ushort_t*)cv((size_t)BGRAPHS * GPAD * 2);
  float*    agg    = (float*)cv((size_t)NNODES * 64 * 4);
  float*    gsin   = (float*)cv((size_t)BGRAPHS * 832 * 4);
  ushort_t* gsinbf = (ushort_t*)cv((size_t)BGRAPHS * 832 * 2);
  ushort_t* gh1bf  = (ushort_t*)cv((size_t)BGRAPHS * GPAD * 2);
  float*    gh2f   = (float*)cv((size_t)BGRAPHS * GPAD * 4);

  for (int i = 0; i < 6; ++i) prep_mlp(pp[i], w[i], stream);

  float* out = (float*)d_out;
  float* e2  = out;
  float* n2  = out + (size_t)NEDGES * 64;
  float* g2  = n2 + (size_t)NNODES * 64;
  float* muo = g2 + (size_t)BGRAPHS * GLOBD;   // mu then logvar, both ones

  const dim3 gg(BGRAPHS / 16, 44);

  // ================= block 1 (globals == ones, folded into bias) =================
  mlp64_fused<0><<<NEDGES / 64, 128, 0, stream>>>(6, edges, nodes, senders, receivers,
      edge_gid, nullptr, w[0].W1sw, w[0].b1ones, w[0].W2sw, pp[0].b2, pp[0].ln_g, pp[0].ln_b, ebuf);
  fill_f32<<<((long)NNODES * 64 + 255) / 256, 256, 0, stream>>>(agg, 0.f, (long)NNODES * 64);
  scatter_add64<<<NEDGES * 64 / 256, 256, 0, stream>>>(ebuf, receivers, agg);
  mlp64_fused<1><<<NNODES / 64, 128, 0, stream>>>(4, agg, nodes, nullptr, nullptr,
      node_gid, nullptr, w[2].W1sw, w[2].b1ones, w[2].W2sw, pp[2].b2, pp[2].ln_g, pp[2].ln_b, nbuf);
  fill_f32<<<(BGRAPHS * 832 + 255) / 256, 256, 0, stream>>>(gsin, 0.f, BGRAPHS * 832);
  segsum64<<<NEDGES / 1024, 64, 0, stream>>>(ebuf, edge_gid, NEDGES, gsin + 0, 832);
  segsum64<<<NNODES / 1024, 64, 0, stream>>>(nbuf, node_gid, NNODES, gsin + 64, 832);
  build_gsin_bf<<<(BGRAPHS * 832 + 255) / 256, 256, 0, stream>>>(gsin, nullptr, gsinbf);
  fill_f32<<<(BGRAPHS * GPAD / 2 + 255) / 256, 256, 0, stream>>>((float*)gh1bf, 0.f, BGRAPHS * GPAD / 2);
  gemm_bf16<<<gg, 32, 0, stream>>>(gsinbf, 832, 4, w[1].W1sw, 44, w[1].b1ones, GLOBD, BGRAPHS,
                                   nullptr, 0, gh1bf, GPAD);
  gemm_bf16<<<gg, 32, 0, stream>>>(gh1bf, GPAD, w[1].KC2, w[1].W2sw, 44, pp[1].b2, GLOBD, BGRAPHS,
                                   gh2f, GPAD, nullptr, 0);
  ln_glob<<<BGRAPHS, 256, 0, stream>>>(gh2f, GPAD, pp[1].ln_g, pp[1].ln_b, gbuf, GLOBD);

  // ================= block 2 (globals = g1) =================
  f32_to_bf16_pad<<<((long)BGRAPHS * GPAD + 255) / 256, 256, 0, stream>>>(
      gbuf, GLOBD, gbf, GPAD, (long)BGRAPHS * GPAD);
  mlp64_fused<0><<<NEDGES / 64, 128, 0, stream>>>(28, ebuf, nbuf, senders, receivers,
      edge_gid, gbf, w[3].W1sw, w[3].b1f, w[3].W2sw, pp[3].b2, pp[3].ln_g, pp[3].ln_b, e2);
  fill_f32<<<((long)NNODES * 64 + 255) / 256, 256, 0, stream>>>(agg, 0.f, (long)NNODES * 64);
  scatter_add64<<<NEDGES * 64 / 256, 256, 0, stream>>>(e2, receivers, agg);
  mlp64_fused<1><<<NNODES / 64, 128, 0, stream>>>(26, agg, nbuf, nullptr, nullptr,
      node_gid, gbf, w[5].W1sw, w[5].b1f, w[5].W2sw, pp[5].b2, pp[5].ln_g, pp[5].ln_b, n2);
  fill_f32<<<(BGRAPHS * 832 + 255) / 256, 256, 0, stream>>>(gsin, 0.f, BGRAPHS * 832);
  segsum64<<<NEDGES / 1024, 64, 0, stream>>>(e2, edge_gid, NEDGES, gsin + 0, 832);
  segsum64<<<NNODES / 1024, 64, 0, stream>>>(n2, node_gid, NNODES, gsin + 64, 832);
  build_gsin_bf<<<(BGRAPHS * 832 + 255) / 256, 256, 0, stream>>>(gsin, gbuf, gsinbf);
  fill_f32<<<(BGRAPHS * GPAD / 2 + 255) / 256, 256, 0, stream>>>((float*)gh1bf, 0.f, BGRAPHS * GPAD / 2);
  gemm_bf16<<<gg, 32, 0, stream>>>(gsinbf, 832, 26, w[4].W1sw, 44, w[4].b1f, GLOBD, BGRAPHS,
                                   nullptr, 0, gh1bf, GPAD);
  gemm_bf16<<<gg, 32, 0, stream>>>(gh1bf, GPAD, w[4].KC2, w[4].W2sw, 44, pp[4].b2, GLOBD, BGRAPHS,
                                   gh2f, GPAD, nullptr, 0);
  ln_glob<<<BGRAPHS, 256, 0, stream>>>(gh2f, GPAD, pp[4].ln_g, pp[4].ln_b, g2, GLOBD);

  // binar: sigmoid on e2[:,0] (after downstream consumers used pre-sigmoid e2)
  sigmoid_col0<<<(NEDGES + 255) / 256, 256, 0, stream>>>(e2);
  // mu, logvar = ones
  fill_f32<<<((long)2 * BGRAPHS * GLOBD + 255) / 256, 256, 0, stream>>>(muo, 1.f,
                                                                        (long)2 * BGRAPHS * GLOBD);
}
